// Adder12Bit_23407571764125
// MI455X (gfx1250) — compile-verified, matrix operand order checked by reading, not run on hardware
//
#include <hip/hip_runtime.h>

typedef __attribute__((ext_vector_type(4)))  float    v4f;
typedef __attribute__((ext_vector_type(8)))  float    v8f;
typedef __attribute__((ext_vector_type(16))) _Float16 v16h;

// 12-bit ripple-carry adder == integer add of bit-weighted sums.
// One V_WMMA_F32_16X16X32_F16 computes valA+valB for 16 rows at once:
//   K0..7   = A cols 11..4 (weights 1..128)        [lanes 0-15, VGPR0-3]
//   K8..11  = A cols 3..0  (weights 256..2048)     [lanes 16-31, VGPR0-1]
//   K12..15 = B cols 11..8 (weights 1..8)          [lanes 16-31, VGPR2-3]
//   K16..23 = zeros (weights mirror w[8..15])      [lanes 0-15, VGPR4-7]
//   K24..31 = B cols 7..0  (weights 16..2048)      [lanes 16-31, VGPR4-7]
// All data is exact in f16 (bits 0/1, weights <= 2048); f32 accumulate exact.
// Rolled loop over 8 tiles amortizes lane/weight setup at low reg pressure.
constexpr int TILES = 8;   // 16-row tiles per wave, rolled loop

__global__ __launch_bounds__(256) void
adder12_wmma_kernel(const float* __restrict__ A,
                    const float* __restrict__ B,
                    float* __restrict__ out,
                    int rows)
{
    const int  lane  = threadIdx.x & 31;
    const int  wid   = threadIdx.x >> 5;
    const int  m     = lane & 15;
    const bool lower = (lane < 16);

    const int tile0 = ((blockIdx.x * 8) + wid) * (16 * TILES);
    if (tile0 >= rows) return;
    const int row0 = tile0 + m;

    // Per-lane streaming pointers, bumped by one tile (192 floats) per iter.
    const float* aBase = A + (size_t)row0 * 12;
    const float* bBase = B + (size_t)row0 * 12;
    const float* s1 = lower ? (aBase + 8) : aBase;        // -> elements 0..3 (reversed)
    const float* s2 = lower ? (aBase + 4) : (bBase + 8);  // -> elements 4..7 (reversed)
    const float* s4 = bBase;                              // upper only -> elems 12..15 (+4 -> 8..11)

    // B-operand: every column identical = power-of-two weight vector.
    // w[16..23] == w[8..15] makes this invariant to the B-layout half-split.
    v16h wv;
    wv[0]  = lower ? (_Float16)1.0f    : (_Float16)256.0f;
    wv[1]  = lower ? (_Float16)2.0f    : (_Float16)512.0f;
    wv[2]  = lower ? (_Float16)4.0f    : (_Float16)1024.0f;
    wv[3]  = lower ? (_Float16)8.0f    : (_Float16)2048.0f;
    wv[4]  = lower ? (_Float16)16.0f   : (_Float16)1.0f;
    wv[5]  = lower ? (_Float16)32.0f   : (_Float16)2.0f;
    wv[6]  = lower ? (_Float16)64.0f   : (_Float16)4.0f;
    wv[7]  = lower ? (_Float16)128.0f  : (_Float16)8.0f;
    wv[8]  = lower ? (_Float16)256.0f  : (_Float16)16.0f;
    wv[9]  = lower ? (_Float16)512.0f  : (_Float16)32.0f;
    wv[10] = lower ? (_Float16)1024.0f : (_Float16)64.0f;
    wv[11] = lower ? (_Float16)2048.0f : (_Float16)128.0f;
    wv[12] = lower ? (_Float16)1.0f    : (_Float16)256.0f;
    wv[13] = lower ? (_Float16)2.0f    : (_Float16)512.0f;
    wv[14] = lower ? (_Float16)4.0f    : (_Float16)1024.0f;
    wv[15] = lower ? (_Float16)8.0f    : (_Float16)2048.0f;

    // Output lane roles: lane n<12 writes sum bit n (shift 11-n) of its half's
    // 8 rows; lanes 12..15 all write the carry (identical value & address ->
    // benign duplicate store, keeps the epilogue branch-free / full-EXEC).
    // All 8 store offsets are off + v*stride (32-bit, scale_offset-friendly);
    // off advances once per tile by step = 8*stride + hop.
    const int  n      = m;
    const bool isBit  = (n < 12);
    const unsigned sh = isBit ? (unsigned)(11 - n) : 12u;
    const int  rb     = tile0 + (lower ? 0 : 8);
    int        off    = isBit ? (rb * 12 + n) : (12 * rows + rb);
    const int  stride = isBit ? 12 : 1;     // element advance per row
    const int  step   = isBit ? 192 : 16;   // element advance per 16-row tile

#pragma unroll 1
    for (int t = 0; t < TILES; ++t) {
        v4f q1 = __builtin_nontemporal_load((const v4f*)s1);
        v4f q2 = __builtin_nontemporal_load((const v4f*)s2);
        v4f q3 = {0.0f, 0.0f, 0.0f, 0.0f};
        v4f q4 = {0.0f, 0.0f, 0.0f, 0.0f};
        if (!lower) {
            q3 = __builtin_nontemporal_load((const v4f*)(s4 + 4));
            q4 = __builtin_nontemporal_load((const v4f*)s4);
        }
        s1 += 192; s2 += 192; s4 += 192;

        // A-operand: this lane's 16 K-values in layout order (cols reversed:
        // bit index 11 is the LSB / weight 2^0).
        v16h av;
        av[0]  = (_Float16)q1.w;  av[1]  = (_Float16)q1.z;
        av[2]  = (_Float16)q1.y;  av[3]  = (_Float16)q1.x;
        av[4]  = (_Float16)q2.w;  av[5]  = (_Float16)q2.z;
        av[6]  = (_Float16)q2.y;  av[7]  = (_Float16)q2.x;
        av[8]  = (_Float16)q3.w;  av[9]  = (_Float16)q3.z;
        av[10] = (_Float16)q3.y;  av[11] = (_Float16)q3.x;
        av[12] = (_Float16)q4.w;  av[13] = (_Float16)q4.z;
        av[14] = (_Float16)q4.y;  av[15] = (_Float16)q4.x;

        // D[row, n] = valA(row) + valB(row), exact integer in [0, 8190].
        v8f cz = {0.f, 0.f, 0.f, 0.f, 0.f, 0.f, 0.f, 0.f};
        v8f c = __builtin_amdgcn_wmma_f32_16x16x32_f16(
                    /*neg_a=*/false, av, /*neg_b=*/false, wv,
                    /*c_mod=*/(short)0, cz,
                    /*reuse_a=*/false, /*reuse_b=*/false);

        // C/D layout: VGPR v of lane l holds row (v + 8*(l>=16)).
#pragma unroll
        for (int v = 0; v < 8; ++v) {
            const unsigned s = (unsigned)c[v];          // exact f32 -> u32
            __builtin_nontemporal_store((float)((s >> sh) & 1u),
                                        out + (off + v * stride));
        }
        off += step;
    }
}

extern "C" void kernel_launch(void* const* d_in, const int* in_sizes, int n_in,
                              void* d_out, int out_size, void* d_ws, size_t ws_size,
                              hipStream_t stream) {
    const float* A = (const float*)d_in[0];
    const float* B = (const float*)d_in[1];
    float* out = (float*)d_out;

    const int rows       = in_sizes[0] / 12;           // 4194304
    const int rowsPerBlk = 16 * TILES * 8;             // 1024
    const int blocks     = (rows + rowsPerBlk - 1) / rowsPerBlk;

    adder12_wmma_kernel<<<dim3(blocks), dim3(256), 0, stream>>>(A, B, out, rows);
}
